// C3k2CC_6700148982073
// MI455X (gfx1250) — matrix operand order repaired and checked by reading, not compiled, not run on hardware
//
#include <hip/hip_runtime.h>
#include <cstdint>
#include <cstddef>

// ---------------------------------------------------------------------------
// CDNA5 (gfx1250) types & helpers
// ---------------------------------------------------------------------------
typedef __attribute__((ext_vector_type(16))) __bf16 v16bf;
typedef __attribute__((ext_vector_type(8)))  float  v8f;

struct alignas(16) u32x4 { unsigned int x, y, z, w; };
struct u32x8 { u32x4 lo, hi; };

__device__ __forceinline__ v16bf make_bf16x16(u32x4 lo, u32x4 hi) {
  u32x8 u; u.lo = lo; u.hi = hi;
  return __builtin_bit_cast(v16bf, u);
}

__device__ __forceinline__ u32x4 zero4() { u32x4 z; z.x = z.y = z.z = z.w = 0u; return z; }

// float -> bf16 (round-to-nearest-even)
__device__ __forceinline__ unsigned short f2bf(float f) {
  unsigned int u = __float_as_uint(f);
  unsigned int r = u + 0x7FFFu + ((u >> 16) & 1u);
  return (unsigned short)(r >> 16);
}

__device__ __forceinline__ v8f wmma_bf16(v16bf a, v16bf b, v8f c) {
  // D = A(16x32) * B(32x16) + C, f32 accumulate
  return __builtin_amdgcn_wmma_f32_16x16x32_bf16(false, a, false, b, (short)0, c,
                                                 false, false);
}

// SiLU with hardware v_rcp_f32 instead of IEEE divide
__device__ __forceinline__ float silu_f(float x) {
  return x * __builtin_amdgcn_rcpf(1.0f + __expf(-x));
}

struct Parts {
  const float* p[4];
  long long bs[4];   // batch stride (floats) per part
};

#define QKV_BS ((size_t)160 * 4096)   // batch stride of fused q|k|v tensor

// ---------------------------------------------------------------------------
// Weight prepack: fold BN scale into weights, emit bf16 W + f32 bias
// ---------------------------------------------------------------------------
__global__ void pack_weights_kernel(const float* __restrict__ w,
                                    const float* __restrict__ g,
                                    const float* __restrict__ bb,
                                    const float* __restrict__ mm,
                                    const float* __restrict__ vv,
                                    const float* __restrict__ bin,
                                    unsigned short* __restrict__ wout,
                                    float* __restrict__ bout,
                                    int Co, int K, int isBN) {
  int i = blockIdx.x * 256 + threadIdx.x;
  if (i < Co * K) {
    float s = 1.0f;
    if (isBN) { int co = i / K; s = g[co] * rsqrtf(vv[co] + 1e-5f); }
    wout[i] = f2bf(w[i] * s);
  }
  if (i < Co) {
    if (isBN) { float s = g[i] * rsqrtf(vv[i] + 1e-5f); bout[i] = bb[i] - mm[i] * s; }
    else bout[i] = bin[i];
  }
}

// ---------------------------------------------------------------------------
// 1x1 conv as GEMM: out[b,co,s] = sum_ci Wp[co,ci]*X[b,ci,s] + bias; opt. SiLU
// Block: 256 thr (8 waves), tile 64(co) x 128(s), K-step 32 via LDS bf16.
// ---------------------------------------------------------------------------
__global__ __launch_bounds__(256) void conv1x1_kernel(
    Parts parts, const unsigned short* __restrict__ Wp,
    const float* __restrict__ bias, float* __restrict__ out,
    int Ci, int Co, int act) {
  const int tid = threadIdx.x, lane = tid & 31, wv = tid >> 5;
  const int waveM = wv >> 1, waveN = wv & 1;
  const int b  = blockIdx.z;
  const int s0 = blockIdx.x * 128;
  const int co0 = blockIdx.y * 64 + waveM * 16;
  __shared__ alignas(16) unsigned short Xs[4][128][8];

  v8f acc[4];
#pragma unroll
  for (int i = 0; i < 4; i++) acc[i] = v8f{0.f,0.f,0.f,0.f,0.f,0.f,0.f,0.f};

  const int l15 = lane & 15;
  const int ka  = (lane < 16) ? 0 : 8;    // A: K groups {ka..ka+7, ka+16..ka+23}
  const int kbB = (lane < 16) ? 0 : 8;    // B: K groups {kbB..kbB+7, kbB+16..kbB+23}
  const bool doC = (co0 < Co);
  const int mrow = co0 + l15;

  for (int k0 = 0; k0 < Ci; k0 += 32) {
#pragma unroll
    for (int i = 0; i < 16; i++) {
      int idx = i * 256 + tid;
      int kk = idx >> 7;           // 0..31
      int s  = idx & 127;
      int kg = k0 + kk;
      int part = kg >> 7;
      const float* src = parts.p[part] + (size_t)b * (size_t)parts.bs[part];
      float val = src[(size_t)(kg & 127) * 4096 + (size_t)(s0 + s)];
      Xs[kk >> 3][s][kk & 7] = f2bf(val);
    }
    __syncthreads();
    if (doC) {
      const unsigned short* wr = Wp + (size_t)mrow * Ci + k0 + ka;
      __builtin_prefetch((const void*)(wr + 32), 0, 1);  // next K-step weights
      v16bf A = make_bf16x16(*(const u32x4*)(wr), *(const u32x4*)(wr + 16));
#pragma unroll
      for (int nt = 0; nt < 4; nt++) {
        int n = waveN * 64 + nt * 16 + l15;
        v16bf B = make_bf16x16(*(const u32x4*)&Xs[kbB >> 3][n][0],
                               *(const u32x4*)&Xs[(kbB >> 3) + 2][n][0]);
        acc[nt] = wmma_bf16(A, B, acc[nt]);
      }
    }
    __syncthreads();
  }

  if (doC) {
#pragma unroll
    for (int nt = 0; nt < 4; nt++) {
      int n = s0 + waveN * 64 + nt * 16 + l15;
#pragma unroll
      for (int r = 0; r < 8; r++) {
        int co = co0 + r + ((lane >= 16) ? 8 : 0);
        float vsum = acc[nt][r] + bias[co];
        if (act) vsum = silu_f(vsum);
        out[((size_t)b * Co + co) * 4096 + n] = vsum;
      }
    }
  }
}

// ---------------------------------------------------------------------------
// 3x3 conv (SAME) as implicit GEMM, K = Cin*9, always SiLU epilogue.
// ---------------------------------------------------------------------------
__global__ __launch_bounds__(256) void conv3x3_kernel(
    const float* __restrict__ in, const unsigned short* __restrict__ Wp,
    const float* __restrict__ bias, float* __restrict__ out, int Cin, int Co) {
  const int Kdim = Cin * 9;
  const int tid = threadIdx.x, lane = tid & 31, wv = tid >> 5;
  const int waveM = wv >> 1, waveN = wv & 1;
  const int b  = blockIdx.z;
  const int s0 = blockIdx.x * 128;
  const int co0 = blockIdx.y * 64 + waveM * 16;
  __shared__ alignas(16) unsigned short Xs[4][128][8];

  v8f acc[4];
#pragma unroll
  for (int i = 0; i < 4; i++) acc[i] = v8f{0.f,0.f,0.f,0.f,0.f,0.f,0.f,0.f};

  const int l15 = lane & 15;
  const int ka  = (lane < 16) ? 0 : 8;
  const int kbB = (lane < 16) ? 0 : 8;
  const bool doC = (co0 < Co);
  const int mrow = co0 + l15;

  for (int k0 = 0; k0 < Kdim; k0 += 32) {
#pragma unroll
    for (int i = 0; i < 16; i++) {
      int idx = i * 256 + tid;
      int kk = idx >> 7;
      int s  = idx & 127;
      int kg = k0 + kk;
      int ci  = kg / 9;
      int tap = kg - ci * 9;
      int sg = s0 + s;
      int hh = (sg >> 6) + (tap / 3) - 1;
      int ww = (sg & 63) + (tap % 3) - 1;
      float val = 0.0f;
      if ((unsigned)hh < 64u && (unsigned)ww < 64u)
        val = in[((size_t)b * Cin + ci) * 4096 + hh * 64 + ww];
      Xs[kk >> 3][s][kk & 7] = f2bf(val);
    }
    __syncthreads();
    if (doC) {
      const unsigned short* wr = Wp + (size_t)mrow * Kdim + k0 + ka;
      __builtin_prefetch((const void*)(wr + 32), 0, 1);
      v16bf A = make_bf16x16(*(const u32x4*)(wr), *(const u32x4*)(wr + 16));
#pragma unroll
      for (int nt = 0; nt < 4; nt++) {
        int n = waveN * 64 + nt * 16 + l15;
        v16bf B = make_bf16x16(*(const u32x4*)&Xs[kbB >> 3][n][0],
                               *(const u32x4*)&Xs[(kbB >> 3) + 2][n][0]);
        acc[nt] = wmma_bf16(A, B, acc[nt]);
      }
    }
    __syncthreads();
  }

  if (doC) {
#pragma unroll
    for (int nt = 0; nt < 4; nt++) {
      int n = s0 + waveN * 64 + nt * 16 + l15;
#pragma unroll
      for (int r = 0; r < 8; r++) {
        int co = co0 + r + ((lane >= 16) ? 8 : 0);
        float vsum = acc[nt][r] + bias[co];
        out[((size_t)b * Co + co) * 4096 + n] = silu_f(vsum);
      }
    }
  }
}

// ---------------------------------------------------------------------------
// CC pass 1: per (b, w-column): eH[h,H] = sum_c q[c,h,w]*k[c,H,w], diag=-inf.
// q at qkv+c*4096, k at qkv+(16+c)*4096. Writes energy layout (b,h,w,H) f32.
// ---------------------------------------------------------------------------
__global__ __launch_bounds__(256) void cc_col_energy(
    const float* __restrict__ qkv, float* __restrict__ energy) {
  const int tid = threadIdx.x, lane = tid & 31, wv = tid >> 5;
  const int b = blockIdx.y, w = blockIdx.x;
  __shared__ alignas(16) unsigned short Qs[64][16];
  __shared__ alignas(16) unsigned short Ks[64][16];
  const float* base = qkv + (size_t)b * QKV_BS;
#pragma unroll
  for (int i = 0; i < 4; i++) {
    int idx = i * 256 + tid;
    int c = idx >> 6, hh = idx & 63;
    Qs[hh][c] = f2bf(base[(size_t)c * 4096 + hh * 64 + w]);
    Ks[hh][c] = f2bf(base[(size_t)(16 + c) * 4096 + hh * 64 + w]);
  }
  __syncthreads();

  const int l15 = lane & 15;
  const int mt = wv >> 1, nh = wv & 1;
  const int m = mt * 16 + l15;
  // K=16 real, pad K 16..31 with zeros
  v16bf A;
  if (lane < 16) A = make_bf16x16(*(const u32x4*)&Qs[m][0], zero4());
  else           A = make_bf16x16(*(const u32x4*)&Qs[m][8], zero4());
#pragma unroll
  for (int j = 0; j < 2; j++) {
    int nt = nh * 2 + j;
    int n = nt * 16 + l15;
    v16bf B;
    if (lane < 16) B = make_bf16x16(*(const u32x4*)&Ks[n][0], zero4());
    else           B = make_bf16x16(*(const u32x4*)&Ks[n][8], zero4());
    v8f acc = v8f{0.f,0.f,0.f,0.f,0.f,0.f,0.f,0.f};
    acc = wmma_bf16(A, B, acc);
#pragma unroll
    for (int r = 0; r < 8; r++) {
      int h = mt * 16 + r + ((lane >= 16) ? 8 : 0);
      int H = nt * 16 + l15;
      float e = acc[r];
      if (h == H) e = -__builtin_inff();
      energy[(((size_t)b * 64 + h) * 64 + w) * 64 + H] = e;
    }
  }
}

// ---------------------------------------------------------------------------
// CC pass 2: per (b, h-row): eW WMMA -> fused 128-wide softmax over [eH|eW]
// -> aH (bf16, layout (b,w,h,H)) and oW = V_row * aW^T -> acc buffer.
// ---------------------------------------------------------------------------
__global__ __launch_bounds__(256) void cc_row_softmax_ow(
    const float* __restrict__ qkv, const float* __restrict__ energy,
    unsigned short* __restrict__ aH, float* __restrict__ accOut) {
  const int tid = threadIdx.x, lane = tid & 31, wv = tid >> 5;
  const int b = blockIdx.y, h = blockIdx.x;
  __shared__ alignas(16) unsigned short Qs[64][16];
  __shared__ alignas(16) unsigned short Ks[64][16];
  __shared__ alignas(16) unsigned short Vs[128][64];
  __shared__ alignas(16) float ews[64][64];
  __shared__ alignas(16) unsigned short aws[64][64];

  const float* base = qkv + (size_t)b * QKV_BS + (size_t)h * 64;
#pragma unroll
  for (int i = 0; i < 4; i++) {
    int idx = i * 256 + tid;
    int c = idx >> 6, ww = idx & 63;
    Qs[ww][c] = f2bf(base[(size_t)c * 4096 + ww]);
    Ks[ww][c] = f2bf(base[(size_t)(16 + c) * 4096 + ww]);
  }
#pragma unroll
  for (int i = 0; i < 32; i++) {
    int idx = i * 256 + tid;
    int c = idx >> 6, ww = idx & 63;
    Vs[c][ww] = f2bf(base[(size_t)(32 + c) * 4096 + ww]);
  }
  __syncthreads();

  const int l15 = lane & 15;
  {  // eW = Q_row^T K_row (64x64, K=16 padded)
    const int mt = wv >> 1, nh_ = wv & 1;
    const int m = mt * 16 + l15;
    v16bf A;
    if (lane < 16) A = make_bf16x16(*(const u32x4*)&Qs[m][0], zero4());
    else           A = make_bf16x16(*(const u32x4*)&Qs[m][8], zero4());
#pragma unroll
    for (int j = 0; j < 2; j++) {
      int nt = nh_ * 2 + j;
      int n = nt * 16 + l15;
      v16bf B;
      if (lane < 16) B = make_bf16x16(*(const u32x4*)&Ks[n][0], zero4());
      else           B = make_bf16x16(*(const u32x4*)&Ks[n][8], zero4());
      v8f acc = v8f{0.f,0.f,0.f,0.f,0.f,0.f,0.f,0.f};
      acc = wmma_bf16(A, B, acc);
#pragma unroll
      for (int r = 0; r < 8; r++) {
        int row = mt * 16 + r + ((lane >= 16) ? 8 : 0);
        ews[row][nt * 16 + l15] = acc[r];
      }
    }
  }
  __syncthreads();

  // softmax over 128 = [eH(global) | eW(LDS)] per output position w = tid
  if (tid < 64) {
    const float* eh = energy + (((size_t)b * 64 + h) * 64 + tid) * 64;
    float mx = -3.0e38f;
    for (int j = 0; j < 64; j++) { float e = eh[j];       if (e > mx) mx = e; }
    for (int j = 0; j < 64; j++) { float e = ews[tid][j]; if (e > mx) mx = e; }
    float sum = 0.f;
    for (int j = 0; j < 64; j++) sum += __expf(eh[j] - mx);
    for (int j = 0; j < 64; j++) { float p = __expf(ews[tid][j] - mx); ews[tid][j] = p; sum += p; }
    float inv = __builtin_amdgcn_rcpf(sum);
    unsigned short* aHrow = aH + (((size_t)b * 64 + tid) * 64 + h) * 64;
    for (int j = 0; j < 64; j++) aHrow[j] = f2bf(__expf(eh[j] - mx) * inv);
    for (int j = 0; j < 64; j++) aws[tid][j] = f2bf(ews[tid][j] * inv);
  }
  __syncthreads();

  // oW[c,w] = sum_W' Vs[c][W'] * aws[w][W'] : M=128, N=64, K=64
  const int c0 = wv * 16;
  const int m = c0 + l15;
  const int ka  = (lane < 16) ? 0 : 8;
  const int kbB = (lane < 16) ? 0 : 8;
  v8f acc[4];
#pragma unroll
  for (int i = 0; i < 4; i++) acc[i] = v8f{0.f,0.f,0.f,0.f,0.f,0.f,0.f,0.f};
#pragma unroll
  for (int k0 = 0; k0 < 64; k0 += 32) {
    v16bf A = make_bf16x16(*(const u32x4*)&Vs[m][k0 + ka],
                           *(const u32x4*)&Vs[m][k0 + ka + 16]);
#pragma unroll
    for (int nt = 0; nt < 4; nt++) {
      int n = nt * 16 + l15;
      v16bf B = make_bf16x16(*(const u32x4*)&aws[n][k0 + kbB],
                             *(const u32x4*)&aws[n][k0 + kbB + 16]);
      acc[nt] = wmma_bf16(A, B, acc[nt]);
    }
  }
#pragma unroll
  for (int nt = 0; nt < 4; nt++) {
#pragma unroll
    for (int r = 0; r < 8; r++) {
      int c = c0 + r + ((lane >= 16) ? 8 : 0);
      int ww = nt * 16 + l15;
      accOut[(((size_t)b * 128 + c) * 64 + h) * 64 + ww] = acc[nt][r];
    }
  }
}

// ---------------------------------------------------------------------------
// CC pass 3: per (b, w-column): oH = V_col * aH^T; out = gamma*(oH+oW)+x1 (+x)
// aH tile (8KB, bf16, contiguous) is staged with GLOBAL_LOAD_ASYNC_TO_LDS_B128
// (CDNA5 async DMA path, tracked by ASYNCcnt).
// ---------------------------------------------------------------------------
__global__ __launch_bounds__(256) void cc_col_oh_out(
    const float* __restrict__ qkv, const unsigned short* __restrict__ aH,
    const float* __restrict__ accIn, const float* __restrict__ x1in,
    const float* __restrict__ extra, long long extraBS, int hasExtra,
    const float* __restrict__ gammaPtr, float* __restrict__ out) {
  const int tid = threadIdx.x, lane = tid & 31, wv = tid >> 5;
  const int b = blockIdx.y, w = blockIdx.x;
  __shared__ alignas(16) unsigned short Vs[128][64];
  __shared__ alignas(16) unsigned short Bs[64][64];

  // Async copy aH[b][w][0:64][0:64] (8KB contiguous bf16) straight into LDS.
  {
    const char* gsrc = (const char*)(aH + ((size_t)b * 64 + w) * 4096);
    unsigned ldsBase = (unsigned)(uintptr_t)(&Bs[0][0]);
#pragma unroll
    for (int i = 0; i < 2; i++) {
      int chunk = i * 256 + tid;                 // 512 x 16B = 8KB
      unsigned ldst = ldsBase + chunk * 16;
      unsigned long long gaddr = (unsigned long long)(gsrc + chunk * 16);
      asm volatile("global_load_async_to_lds_b128 %0, %1, off"
                   :: "v"(ldst), "v"(gaddr) : "memory");
    }
  }

  const float* base = qkv + (size_t)b * QKV_BS + (size_t)(32) * 4096 + (size_t)w;
#pragma unroll
  for (int i = 0; i < 32; i++) {
    int idx = i * 256 + tid;
    int c = idx >> 6, H = idx & 63;
    Vs[c][H] = f2bf(base[(size_t)c * 4096 + (size_t)H * 64]);
  }
  asm volatile("s_wait_asynccnt 0x0" ::: "memory");
  __syncthreads();

  const float gamma = gammaPtr[0];
  const int l15 = lane & 15;
  const int ka  = (lane < 16) ? 0 : 8;
  const int kbB = (lane < 16) ? 0 : 8;
  const int c0 = wv * 16;
  const int m = c0 + l15;
  v8f acc[4];
#pragma unroll
  for (int i = 0; i < 4; i++) acc[i] = v8f{0.f,0.f,0.f,0.f,0.f,0.f,0.f,0.f};
#pragma unroll
  for (int k0 = 0; k0 < 64; k0 += 32) {
    v16bf A = make_bf16x16(*(const u32x4*)&Vs[m][k0 + ka],
                           *(const u32x4*)&Vs[m][k0 + ka + 16]);
#pragma unroll
    for (int nt = 0; nt < 4; nt++) {
      int n = nt * 16 + l15;
      v16bf B = make_bf16x16(*(const u32x4*)&Bs[n][k0 + kbB],
                             *(const u32x4*)&Bs[n][k0 + kbB + 16]);
      acc[nt] = wmma_bf16(A, B, acc[nt]);
    }
  }
#pragma unroll
  for (int nt = 0; nt < 4; nt++) {
#pragma unroll
    for (int r = 0; r < 8; r++) {
      int c  = c0 + r + ((lane >= 16) ? 8 : 0);
      int hh = nt * 16 + l15;
      size_t off = ((size_t)b * 128 + c) * 4096 + (size_t)hh * 64 + w;
      float o = gamma * (acc[nt][r] + accIn[off]) + x1in[off];
      if (hasExtra)
        o += extra[(size_t)b * (size_t)extraBS + (size_t)c * 4096 + (size_t)hh * 64 + w];
      out[off] = o;
    }
  }
}

// ---------------------------------------------------------------------------
// Host-side orchestration
// ---------------------------------------------------------------------------
extern "C" void kernel_launch(void* const* d_in, const int* in_sizes, int n_in,
                              void* d_out, int out_size, void* d_ws, size_t ws_size,
                              hipStream_t stream) {
  (void)in_sizes; (void)n_in; (void)out_size; (void)ws_size;
  // Input order (setup_inputs dict order):
  // 0: x
  // 1..5:  cv1 {w, bn_g, bn_b, bn_m, bn_v}
  // 6..10: cv2 {w, bn_g, bn_b, bn_m, bn_v}
  // 11..27:  m[0] {cv1(5), cv2(5), q_w,q_b,k_w,k_b,v_w,v_b,gamma}
  // 28..44:  m[1] same
  auto F = [&](int i) { return (const float*)d_in[i]; };
  const float* X = F(0);

  char* ws = (char*)d_ws;
  size_t off = 0;
  auto alloc = [&](size_t bytes) -> char* {
    char* p = ws + off;
    off = (off + bytes + 255) & ~(size_t)255;
    return p;
  };

  unsigned short* wpA = (unsigned short*)alloc((size_t)256 * 256 * 2);
  float* bA = (float*)alloc(256 * 4);
  unsigned short* wpF = (unsigned short*)alloc((size_t)256 * 512 * 2);
  float* bF = (float*)alloc(256 * 4);
  unsigned short *wp1[2], *wp2[2], *wqkv[2];
  float *b1[2], *b2[2], *bqkv[2];
  for (int i = 0; i < 2; i++) {
    wp1[i]  = (unsigned short*)alloc((size_t)64 * 128 * 2);   b1[i]  = (float*)alloc(64 * 4);
    wp2[i]  = (unsigned short*)alloc((size_t)128 * 576 * 2);  b2[i]  = (float*)alloc(128 * 4);
    wqkv[i] = (unsigned short*)alloc((size_t)160 * 128 * 2);  bqkv[i] = (float*)alloc(160 * 4);
  }
  float* yA  = (float*)alloc((size_t)16 * 256 * 4096 * 4);
  float* y23[2];
  y23[0] = (float*)alloc((size_t)16 * 128 * 4096 * 4);
  y23[1] = (float*)alloc((size_t)16 * 128 * 4096 * 4);
  float* t1   = (float*)alloc((size_t)16 * 64 * 4096 * 4);
  float* x1a  = (float*)alloc((size_t)16 * 128 * 4096 * 4);
  float* x1b  = (float*)alloc((size_t)16 * 128 * 4096 * 4);
  float* qkvb = (float*)alloc((size_t)16 * 160 * 4096 * 4);
  float* energy = (float*)alloc((size_t)16 * 64 * 64 * 64 * 4);
  unsigned short* aHb = (unsigned short*)alloc((size_t)16 * 64 * 64 * 64 * 2);
  float* accb = (float*)alloc((size_t)16 * 128 * 4096 * 4);

  auto pack = [&](const float* w, const float* g, const float* bb, const float* mm,
                  const float* vv, const float* bin, unsigned short* wout, float* bout,
                  int Co, int K, int isBN) {
    int tot = Co * K;
    pack_weights_kernel<<<dim3((tot + 255) / 256), dim3(256), 0, stream>>>(
        w, g, bb, mm, vv, bin, wout, bout, Co, K, isBN);
  };
  pack(F(1), F(2), F(3), F(4), F(5), nullptr, wpA, bA, 256, 256, 1);
  pack(F(6), F(7), F(8), F(9), F(10), nullptr, wpF, bF, 256, 512, 1);
  for (int i = 0; i < 2; i++) {
    int base = 11 + i * 17;
    pack(F(base + 0), F(base + 1), F(base + 2), F(base + 3), F(base + 4), nullptr,
         wp1[i], b1[i], 64, 128, 1);
    pack(F(base + 5), F(base + 6), F(base + 7), F(base + 8), F(base + 9), nullptr,
         wp2[i], b2[i], 128, 576, 1);
    // fused q|k|v weights: rows [0:16)=q, [16:32)=k, [32:160)=v
    pack(F(base + 10), nullptr, nullptr, nullptr, nullptr, F(base + 11),
         wqkv[i] + 0 * 128, bqkv[i] + 0, 16, 128, 0);
    pack(F(base + 12), nullptr, nullptr, nullptr, nullptr, F(base + 13),
         wqkv[i] + 16 * 128, bqkv[i] + 16, 16, 128, 0);
    pack(F(base + 14), nullptr, nullptr, nullptr, nullptr, F(base + 15),
         wqkv[i] + 32 * 128, bqkv[i] + 32, 128, 128, 0);
  }

  auto conv1x1 = [&](Parts P, const unsigned short* Wp, const float* bias, float* out,
                     int Ci, int Co, int act) {
    conv1x1_kernel<<<dim3(32, (Co + 63) / 64, 16), dim3(256), 0, stream>>>(
        P, Wp, bias, out, Ci, Co, act);
  };
  auto mk1 = [&](const float* p0, long long bs) {
    Parts P;
    P.p[0] = P.p[1] = P.p[2] = P.p[3] = p0;
    P.bs[0] = P.bs[1] = P.bs[2] = P.bs[3] = bs;
    return P;
  };

  // --- cv1 main: 256 -> 256, SiLU ---
  {
    Parts P;
    P.p[0] = X; P.p[1] = X + (size_t)128 * 4096; P.p[2] = X; P.p[3] = X;
    P.bs[0] = P.bs[1] = P.bs[2] = P.bs[3] = (long long)256 * 4096;
    conv1x1(P, wpA, bA, yA, 256, 256, 1);
  }

  const float* curSrc = yA + (size_t)128 * 4096;  // y[:,128:]
  long long curBS = (long long)256 * 4096;
  for (int i = 0; i < 2; i++) {
    const float* gamma = F(11 + i * 17 + 16);
    // cv1 of bottleneck: 128 -> 64 SiLU
    conv1x1(mk1(curSrc, curBS), wp1[i], b1[i], t1, 128, 64, 1);
    // cv2: 3x3, 64 -> 128, SiLU
    conv3x3_kernel<<<dim3(32, 2, 16), dim3(256), 0, stream>>>(t1, wp2[i], b2[i], x1a, 64, 128);

    // --- criss-cross recurrence r = 0: x1a -> x1b ---
    {
      conv1x1(mk1(x1a, (long long)128 * 4096), wqkv[i], bqkv[i], qkvb, 128, 160, 0);
      cc_col_energy<<<dim3(64, 16), dim3(256), 0, stream>>>(qkvb, energy);
      cc_row_softmax_ow<<<dim3(64, 16), dim3(256), 0, stream>>>(qkvb, energy, aHb, accb);
      cc_col_oh_out<<<dim3(64, 16), dim3(256), 0, stream>>>(
          qkvb, aHb, accb, x1a, (const float*)nullptr, 0LL, 0, gamma, x1b);
    }
    // --- recurrence r = 1: x1b -> y23[i], fused +x residual ---
    {
      conv1x1(mk1(x1b, (long long)128 * 4096), wqkv[i], bqkv[i], qkvb, 128, 160, 0);
      cc_col_energy<<<dim3(64, 16), dim3(256), 0, stream>>>(qkvb, energy);
      cc_row_softmax_ow<<<dim3(64, 16), dim3(256), 0, stream>>>(qkvb, energy, aHb, accb);
      cc_col_oh_out<<<dim3(64, 16), dim3(256), 0, stream>>>(
          qkvb, aHb, accb, x1b, curSrc, curBS, 1, gamma, y23[i]);
    }
    curSrc = y23[i];
    curBS = (long long)128 * 4096;
  }

  // --- final cv2: concat(y0,y1,y2,y3) 512 -> 256, SiLU -> d_out ---
  {
    Parts P;
    P.p[0] = yA;
    P.p[1] = yA + (size_t)128 * 4096;
    P.p[2] = y23[0];
    P.p[3] = y23[1];
    P.bs[0] = P.bs[1] = (long long)256 * 4096;
    P.bs[2] = P.bs[3] = (long long)128 * 4096;
    conv1x1(P, wpF, bF, (float*)d_out, 512, 256, 1);
  }
}
